// GNNEncoder_10617159156305
// MI455X (gfx1250) — matrix operand order repaired and checked by reading, not compiled, not run on hardware
//
#include <hip/hip_runtime.h>

#define NN 50000
#define NE 150000
#define HF 512
#define NHID 64
#define NG 64

typedef __attribute__((ext_vector_type(16))) _Float16 v16h;
typedef __attribute__((ext_vector_type(8)))  float    v8f;

static __device__ __forceinline__ float leakyf(float v){ return v > 0.f ? v : 0.01f * v; }

// =====================================================================
// WMMA GEMM: C[M,N] = act(A[M,K] @ W[K,N] + bias[N]) (* mask[row] optional)
// f32 in global, converted to f16 while staging into LDS, f32 accumulate.
// Block: 256 threads (8 waves), tile 128(M) x 64(N), K-step 32.
// Waves arranged 4(M) x 2(N); each wave computes 32x32 = 2x2 WMMA tiles.
// Software-pipelined double buffer: global loads for tile k+1 issue before
// the WMMAs of tile k; one barrier per K-step.
// A-row indices are CLAMPED (not guarded): LDS row r only feeds output row
// bm+r, which is guarded at the store, so no divergent exec-mask staging.
// =====================================================================
#define TM 128
#define TN 64
#define TK 32
#define LSTR 34   // padded f16 stride (17 dwords) -> conflict-free-ish banks

__global__ __launch_bounds__(256) void k_gemm_wmma(
    const float* __restrict__ A, const float* __restrict__ W,
    const float* __restrict__ bias, float* __restrict__ C,
    const float* __restrict__ mask, int M, int N, int K, int act)
{
  __shared__ _Float16 lA[2][TM][LSTR];
  __shared__ _Float16 lB[2][TN][LSTR];   // W tile transposed: [n][k]
  const int tid  = threadIdx.x;
  const int bm   = blockIdx.x * TM;
  const int bn   = blockIdx.y * TN;
  const int wave = tid >> 5;
  const int lane = tid & 31;
  const int wm   = (wave & 3) * 32;   // wave M offset inside block tile
  const int wn   = (wave >> 2) * 32;  // wave N offset inside block tile
  const int half = lane >> 4;         // K-half / M-half selector
  const int lr   = lane & 15;

  // staging register state (per thread): 4 float4 of A, 8 floats of W
  float4 avals[4];
  float  wvals[8];
  int    arow[4];
  int    ac4[4];
  int    wkk[8], wnn[8];
#pragma unroll
  for (int i = 0; i < 4; ++i) {
    int idx = tid + i * 256;          // 0..1023 float4 slots
    int r   = idx >> 3;               // 8 float4 per row
    ac4[i]  = (idx & 7) << 2;
    int grow = bm + r;
    if (grow >= M) grow = M - 1;      // clamp (safe: see header comment)
    arow[i] = r;
    // stash absolute row for address computation
    avals[i] = {0.f, 0.f, 0.f, 0.f};
    // store absolute row in arow? we need both LDS row r and global row.
    // keep LDS row in arow[], recompute global row below via bm+... clamp:
    // simpler: precompute pointer offsets
    (void)grow;
  }
#pragma unroll
  for (int i = 0; i < 8; ++i) {
    int idx = tid + i * 256;          // 0..2047
    wkk[i] = idx >> 6;                // 0..31
    wnn[i] = idx & 63;
  }

  auto gload = [&](int k0) {
#pragma unroll
    for (int i = 0; i < 4; ++i) {
      int grow = bm + arow[i];
      if (grow >= M) grow = M - 1;
      avals[i] = *(const float4*)(A + (size_t)grow * K + k0 + ac4[i]);
    }
#pragma unroll
    for (int i = 0; i < 8; ++i)
      wvals[i] = W[(size_t)(k0 + wkk[i]) * N + bn + wnn[i]];
  };
  auto sstore = [&](int buf) {
#pragma unroll
    for (int i = 0; i < 4; ++i) {
      _Float16* dstp = &lA[buf][arow[i]][ac4[i]];
      dstp[0] = (_Float16)avals[i].x;
      dstp[1] = (_Float16)avals[i].y;
      dstp[2] = (_Float16)avals[i].z;
      dstp[3] = (_Float16)avals[i].w;
    }
#pragma unroll
    for (int i = 0; i < 8; ++i)
      lB[buf][wnn[i]][wkk[i]] = (_Float16)wvals[i];
  };

  v8f acc[2][2] = {};

  gload(0);
  sstore(0);
  __syncthreads();
  int cur = 0;

  for (int k0 = 0; k0 < K; k0 += TK) {
    const bool more = (k0 + TK) < K;
    if (more) gload(k0 + TK);   // issue next-tile global loads before compute

#pragma unroll
    for (int im = 0; im < 2; ++im) {
      // A fragment: lane holds row (wm+im*16+lr), K selected by half.
      // v16h element pair (2j,2j+1) = consecutive K at kb (ISA 16-bit A layout).
      v16h af;
      const _Float16* ap = &lA[cur][wm + im * 16 + lr][0];
#pragma unroll
      for (int j = 0; j < 8; ++j) {
        int kb = ((j < 4) ? 0 : 16) + half * 8 + (j & 3) * 2;
        af[2 * j]     = ap[kb];
        af[2 * j + 1] = ap[kb + 1];
      }
#pragma unroll
      for (int jn = 0; jn < 2; ++jn) {
        // B fragment: lane holds column (wn+jn*16+lr), K = (e<8?0:16) + half*8 + (e&7).
        v16h bf;
        const _Float16* bp = &lB[cur][wn + jn * 16 + lr][0];
#pragma unroll
        for (int e = 0; e < 16; ++e) bf[e] = bp[((e < 8) ? 0 : 16) + half * 8 + (e & 7)];
        acc[im][jn] = __builtin_amdgcn_wmma_f32_16x16x32_f16(
            false, af, false, bf, (short)0, acc[im][jn], false, false);
      }
    }

    if (more) sstore(cur ^ 1);  // fill alternate buffer (no conflict with readers)
    cur ^= 1;
    __syncthreads();
  }

  // ---- epilogue: C/D layout -> VGPR r: M = half*8 + r, N = lane&15 ----
#pragma unroll
  for (int im = 0; im < 2; ++im) {
#pragma unroll
    for (int jn = 0; jn < 2; ++jn) {
      int col = bn + wn + jn * 16 + lr;
      float bv = bias[col];
#pragma unroll
      for (int r = 0; r < 8; ++r) {
        int row = bm + wm + im * 16 + half * 8 + r;
        if (row < M) {
          float v = acc[im][jn][r] + bv;
          if (act == 1)      v = fmaxf(v, 0.f);
          else if (act == 2) v = leakyf(v);
          if (mask) v *= mask[row];
          C[(size_t)row * N + col] = v;
        }
      }
    }
  }
}

// =====================================================================
// Elementwise / scatter / reduction kernels
// =====================================================================
__global__ void k_counts(const int* __restrict__ batch, float* __restrict__ counts, int n) {
  int i = blockIdx.x * 256 + threadIdx.x;
  if (i < n) atomicAdd(&counts[batch[i]], 1.0f);
}
__global__ void k_invs(const int* __restrict__ batch, const float* __restrict__ counts,
                       float* __restrict__ invs, int n) {
  int i = blockIdx.x * 256 + threadIdx.x;
  if (i < n) invs[i] = rsqrtf(counts[batch[i]]);
}
__global__ void k_mask_init(const float* __restrict__ x, float* __restrict__ m, int n) {
  int i = blockIdx.x * 256 + threadIdx.x;
  if (i < n) m[i] = (fabsf(x[i]) > 0.f) ? 1.f : 0.f;
}
__global__ void k_edge_scatter_1d(const float* __restrict__ val, const int* __restrict__ src,
                                  const int* __restrict__ dst, float* __restrict__ agg, int e) {
  int i = blockIdx.x * 256 + threadIdx.x;
  if (i < e) atomicAdd(&agg[dst[i]], val[src[i]]);
}
__global__ void k_mask_threshold(const float* __restrict__ m, const float* __restrict__ agg,
                                 float* __restrict__ mout, int n) {
  int i = blockIdx.x * 256 + threadIdx.x;
  if (i < n) mout[i] = (m[i] + agg[i] > 0.f) ? 1.f : 0.f;
}
__global__ void k_combine_1d(const float* __restrict__ x, const float* __restrict__ agg,
                             const float* __restrict__ epsp, float* __restrict__ h, int n) {
  int i = blockIdx.x * 256 + threadIdx.x;
  if (i < n) h[i] = (1.0f + epsp[0]) * x[i] + agg[i];
}
__global__ void k_lin_1toH(const float* __restrict__ h, const float* __restrict__ w,
                           const float* __restrict__ b, float* __restrict__ o, int total) {
  int i = blockIdx.x * 256 + threadIdx.x;
  if (i >= total) return;
  int r = i >> 9, c = i & (HF - 1);
  o[i] = fmaxf(h[r] * w[c] + b[c], 0.f);
}
__global__ void k_edge_scatter_H(const float* __restrict__ x, const int* __restrict__ src,
                                 const int* __restrict__ dst, float* __restrict__ agg, int total) {
  int i = blockIdx.x * 256 + threadIdx.x;
  if (i >= total) return;
  int e = i >> 9, f = i & (HF - 1);
  atomicAdd(&agg[(size_t)dst[e] * HF + f], x[(size_t)src[e] * HF + f]);
}
__global__ void k_combine_H(const float* __restrict__ x, const float* __restrict__ agg,
                            const float* __restrict__ epsp, int epsidx,
                            float* __restrict__ h, int total) {
  int i = blockIdx.x * 256 + threadIdx.x;
  if (i >= total) return;
  h[i] = (1.0f + epsp[epsidx]) * x[i] + agg[i];
}

#define BN_ROWS 256
__global__ __launch_bounds__(256) void k_bn_stats(
    const float* __restrict__ X, const float* __restrict__ mask,
    const float* __restrict__ invs, float* __restrict__ sums,
    float* __restrict__ sumsq, int M, int N, int scaled)
{
  int r0 = blockIdx.x * BN_ROWS;
  int rend = r0 + BN_ROWS; if (rend > M) rend = M;
  for (int c = threadIdx.x; c < N; c += 256) {
    float s = 0.f, q = 0.f;
    for (int r = r0; r < rend; ++r) {
      float v = X[(size_t)r * N + c];
      if (scaled) v *= mask[r] * invs[r];
      s += v; q += v * v;
    }
    atomicAdd(&sums[c], s);
    atomicAdd(&sumsq[c], q);
  }
}
__global__ void k_bn_finalize(const float* __restrict__ sums, const float* __restrict__ sumsq,
                              const float* __restrict__ g, const float* __restrict__ b,
                              float* __restrict__ scl, float* __restrict__ shf, int M, int N) {
  int c = blockIdx.x * 256 + threadIdx.x;
  if (c >= N) return;
  float mean = sums[c] / (float)M;
  float var  = sumsq[c] / (float)M - mean * mean;
  float sc   = g[c] * rsqrtf(var + 1e-5f);
  scl[c] = sc;
  shf[c] = b[c] - mean * sc;
}
__global__ void k_bn_apply_leaky(const float* __restrict__ h, const float* __restrict__ scl,
                                 const float* __restrict__ shf, float* __restrict__ x, int total) {
  int i = blockIdx.x * 256 + threadIdx.x;
  if (i >= total) return;
  int c = i & (HF - 1);
  x[i] = leakyf(scl[c] * h[i] + shf[c]);
}
__global__ void k_bn_apply_res_leaky(const float* __restrict__ h, const float* __restrict__ scl,
                                     const float* __restrict__ shf, float* __restrict__ x, int total) {
  int i = blockIdx.x * 256 + threadIdx.x;
  if (i >= total) return;
  int c = i & (HF - 1);
  x[i] += leakyf(scl[c] * h[i] + shf[c]);
}
__global__ void k_bn_apply_scaled(float* __restrict__ x, const float* __restrict__ mask,
                                  const float* __restrict__ invs, const float* __restrict__ scl,
                                  const float* __restrict__ shf, int total) {
  int i = blockIdx.x * 256 + threadIdx.x;
  if (i >= total) return;
  int c = i & (HF - 1), r = i >> 9;
  x[i] = scl[c] * (x[i] * mask[r] * invs[r]) + shf[c];
}

// lin2: one wave per row, 64-dot + leaky + mask
__global__ __launch_bounds__(256) void k_lin2(const float* __restrict__ y1, const float* __restrict__ w,
                                              const float* __restrict__ b, const float* __restrict__ mask,
                                              float* __restrict__ y2, int M) {
  int row  = blockIdx.x * 8 + (threadIdx.x >> 5);
  int lane = threadIdx.x & 31;
  if (row >= M) return;
  float s = y1[(size_t)row * 64 + lane] * w[lane] + y1[(size_t)row * 64 + 32 + lane] * w[32 + lane];
  for (int off = 16; off > 0; off >>= 1) s += __shfl_down(s, off, 32);
  if (lane == 0) {
    float v = leakyf(s + b[0]);
    y2[row] = v * mask[row];
  }
}

static __device__ __forceinline__ void atomicMaxF(float* addr, float val) {
  unsigned* a = (unsigned*)addr;
  unsigned old = *a;
  while (__uint_as_float(old) < val) {
    unsigned assumed = old;
    old = atomicCAS(a, assumed, __float_as_uint(val));
    if (old == assumed) break;
  }
}
static __device__ __forceinline__ void atomicMinF(float* addr, float val) {
  unsigned* a = (unsigned*)addr;
  unsigned old = *a;
  while (__uint_as_float(old) > val) {
    unsigned assumed = old;
    old = atomicCAS(a, assumed, __float_as_uint(val));
    if (old == assumed) break;
  }
}
__global__ void k_minmax_init(float* __restrict__ bmn, float* __restrict__ bmx) {
  int i = threadIdx.x;
  if (i < NG) { bmn[i] = 3.4e38f; bmx[i] = -3.4e38f; }
}
__global__ void k_minmax(const float* __restrict__ y2, const int* __restrict__ batch,
                         float* __restrict__ bmn, float* __restrict__ bmx, int n) {
  int i = blockIdx.x * 256 + threadIdx.x;
  if (i >= n) return;
  int g = batch[i];
  atomicMaxF(&bmx[g], y2[i]);
  atomicMinF(&bmn[g], y2[i]);
}
__global__ void k_final(const float* __restrict__ y2, const int* __restrict__ batch,
                        const float* __restrict__ bmn, const float* __restrict__ bmx,
                        float* __restrict__ out, int n) {
  int i = blockIdx.x * 256 + threadIdx.x;
  if (i >= n) return;
  int g = batch[i];
  float lo = bmn[g], hi = bmx[g];
  out[i] = (y2[i] - lo) / (hi + 1e-6f - lo);
}

// =====================================================================
// Host orchestration
// =====================================================================
static inline int cdiv(int a, int b) { return (a + b - 1) / b; }

extern "C" void kernel_launch(void* const* d_in, const int* in_sizes, int n_in,
                              void* d_out, int out_size, void* d_ws, size_t ws_size,
                              hipStream_t stream)
{
  (void)in_sizes; (void)n_in; (void)out_size; (void)ws_size;

  const float* x_in  = (const float*)d_in[0];
  const int*   ei    = (const int*)d_in[1];
  const int*   batch = (const int*)d_in[2];
  const float* c1w1  = (const float*)d_in[3];
  const float* c1b1  = (const float*)d_in[4];
  const float* c1w2  = (const float*)d_in[5];
  const float* c1b2  = (const float*)d_in[6];
  const float* c1bng = (const float*)d_in[7];
  const float* c1bnb = (const float*)d_in[8];
  const float* eps1  = (const float*)d_in[9];
  const float* bn1g  = (const float*)d_in[10];
  const float* bn1b  = (const float*)d_in[11];
  const float* cw1   = (const float*)d_in[12];
  const float* cb1   = (const float*)d_in[13];
  const float* cw2   = (const float*)d_in[14];
  const float* cb2   = (const float*)d_in[15];
  const float* cbng  = (const float*)d_in[16];
  const float* cbnb  = (const float*)d_in[17];
  const float* ceps  = (const float*)d_in[18];
  const float* bnsg  = (const float*)d_in[19];
  const float* bnsb  = (const float*)d_in[20];
  const float* l1w   = (const float*)d_in[21];
  const float* l1b   = (const float*)d_in[22];
  const float* l2w   = (const float*)d_in[23];
  const float* l2b   = (const float*)d_in[24];
  float* out = (float*)d_out;
  const int* src = ei;
  const int* dst = ei + NE;

  const size_t NH = (size_t)NN * HF;
  float* p    = (float*)d_ws;
  float* xbuf = p;  p += NH;
  float* t1   = p;  p += NH;
  float* t2   = p;  p += NH;
  float* mA   = p;  p += NN;
  float* mB   = p;  p += NN;
  float* invs = p;  p += NN;
  float* maggr= p;  p += NN;
  float* hcol = p;  p += NN;
  float* sums = p;  p += HF;
  float* sumsq= p;  p += HF;
  float* scl  = p;  p += HF;
  float* shf  = p;  p += HF;
  float* cnts = p;  p += NG;
  float* bmn  = p;  p += NG;
  float* bmx  = p;  p += NG;
  float* y1   = t1;    // head reuses big buffers
  float* y2   = hcol;

  const int gN    = cdiv(NN, 256);
  const int gNH   = cdiv((int)NH, 256);
  const int gEH   = cdiv(NE * HF, 256);
  const int gE    = cdiv(NE, 256);
  const int gBN   = cdiv(NN, BN_ROWS);
  dim3 gemmGridH(cdiv(NN, TM), HF / TN);
  dim3 gemmGrid1(cdiv(NN, TM), NHID / TN);

  float* mcur = mA;
  float* mnext = mB;
  auto propagate_mask = [&]() {
    hipMemsetAsync(maggr, 0, NN * sizeof(float), stream);
    k_edge_scatter_1d<<<gE, 256, 0, stream>>>(mcur, src, dst, maggr, NE);
    k_mask_threshold<<<gN, 256, 0, stream>>>(mcur, maggr, mnext, NN);
    float* t = mcur; mcur = mnext; mnext = t;
  };

  // --- counts / inv_sqrt_n ---
  hipMemsetAsync(cnts, 0, NG * sizeof(float), stream);
  k_counts<<<gN, 256, 0, stream>>>(batch, cnts, NN);
  k_invs<<<gN, 256, 0, stream>>>(batch, cnts, invs, NN);

  // --- initial mask + one propagation ---
  k_mask_init<<<gN, 256, 0, stream>>>(x_in, mcur, NN);
  propagate_mask();                     // mask now in mcur

  // --- conv1: GIN on (N,1) then 1->512 -> 512x512 ---
  hipMemsetAsync(maggr, 0, NN * sizeof(float), stream);
  k_edge_scatter_1d<<<gE, 256, 0, stream>>>(x_in, src, dst, maggr, NE);
  k_combine_1d<<<gN, 256, 0, stream>>>(x_in, maggr, eps1, hcol, NN);
  k_lin_1toH<<<gNH, 256, 0, stream>>>(hcol, c1w1, c1b1, t1, (int)NH);
  k_gemm_wmma<<<gemmGridH, 256, 0, stream>>>(t1, c1w2, c1b2, t2, nullptr, NN, HF, HF, 1);
  // conv1 BN + leaky -> xbuf
  hipMemsetAsync(sums, 0, 2 * HF * sizeof(float), stream);
  k_bn_stats<<<gBN, 256, 0, stream>>>(t2, mcur, invs, sums, sumsq, NN, HF, 0);
  k_bn_finalize<<<cdiv(HF, 256), 256, 0, stream>>>(sums, sumsq, c1bng, c1bnb, scl, shf, NN, HF);
  k_bn_apply_leaky<<<gNH, 256, 0, stream>>>(t2, scl, shf, xbuf, (int)NH);
  // bn1 over x*mask*invs, in place
  hipMemsetAsync(sums, 0, 2 * HF * sizeof(float), stream);
  k_bn_stats<<<gBN, 256, 0, stream>>>(xbuf, mcur, invs, sums, sumsq, NN, HF, 1);
  k_bn_finalize<<<cdiv(HF, 256), 256, 0, stream>>>(sums, sumsq, bn1g, bn1b, scl, shf, NN, HF);
  k_bn_apply_scaled<<<gNH, 256, 0, stream>>>(xbuf, mcur, invs, scl, shf, (int)NH);

  // --- 3 middle GIN layers with residual ---
  for (int i = 0; i < 3; ++i) {
    const size_t wOff = (size_t)i * HF * HF;
    const size_t bOff = (size_t)i * HF;
    hipMemsetAsync(t1, 0, NH * sizeof(float), stream);
    k_edge_scatter_H<<<gEH, 256, 0, stream>>>(xbuf, src, dst, t1, NE * HF);
    k_combine_H<<<gNH, 256, 0, stream>>>(xbuf, t1, ceps, i, t2, (int)NH);
    k_gemm_wmma<<<gemmGridH, 256, 0, stream>>>(t2, cw1 + wOff, cb1 + bOff, t1, nullptr, NN, HF, HF, 1);
    k_gemm_wmma<<<gemmGridH, 256, 0, stream>>>(t1, cw2 + wOff, cb2 + bOff, t2, nullptr, NN, HF, HF, 1);
    // conv BN + leaky + residual into xbuf
    hipMemsetAsync(sums, 0, 2 * HF * sizeof(float), stream);
    k_bn_stats<<<gBN, 256, 0, stream>>>(t2, mcur, invs, sums, sumsq, NN, HF, 0);
    k_bn_finalize<<<cdiv(HF, 256), 256, 0, stream>>>(sums, sumsq, cbng + bOff, cbnb + bOff, scl, shf, NN, HF);
    k_bn_apply_res_leaky<<<gNH, 256, 0, stream>>>(t2, scl, shf, xbuf, (int)NH);
    // mask propagation
    propagate_mask();
    // outer BN over x*mask*invs, in place
    hipMemsetAsync(sums, 0, 2 * HF * sizeof(float), stream);
    k_bn_stats<<<gBN, 256, 0, stream>>>(xbuf, mcur, invs, sums, sumsq, NN, HF, 1);
    k_bn_finalize<<<cdiv(HF, 256), 256, 0, stream>>>(sums, sumsq, bnsg + bOff, bnsb + bOff, scl, shf, NN, HF);
    k_bn_apply_scaled<<<gNH, 256, 0, stream>>>(xbuf, mcur, invs, scl, shf, (int)NH);
  }

  // --- head: lin1 (512->64, leaky, *mask) via WMMA, lin2 (64->1) ---
  k_gemm_wmma<<<gemmGrid1, 256, 0, stream>>>(xbuf, l1w, l1b, y1, mcur, NN, NHID, HF, 2);
  k_lin2<<<cdiv(NN, 8), 256, 0, stream>>>(y1, l2w, l2b, mcur, y2, NN);

  // --- per-graph min/max normalize ---
  k_minmax_init<<<1, 64, 0, stream>>>(bmn, bmx);
  k_minmax<<<gN, 256, 0, stream>>>(y2, batch, bmn, bmx, NN);
  k_final<<<gN, 256, 0, stream>>>(y2, batch, bmn, bmx, out, NN);
}
